// PointNetEncoder_29214367548139
// MI455X (gfx1250) — compile-verified, matrix operand order checked by reading, not run on hardware
//
#include <hip/hip_runtime.h>

typedef float     v2f  __attribute__((ext_vector_type(2)));
typedef float     v8f  __attribute__((ext_vector_type(8)));
typedef _Float16  v16h __attribute__((ext_vector_type(16)));

#define R2MAX 0.01f      /* 0.1^2  */
#define R2MID 0.0025f    /* 0.05^2 */
#define R2MIN 0.0004f    /* 0.02^2 */
#define BIGF  3.0e38f

// ---------------------------------------------------------------------------
// Kernel 1: fused WMMA distance tiles + per-row sorted top-64 (d^2 < 0.01).
// 2 waves per block; wave w scans point tiles w, w+2, ... for the same 32
// grid rows into private LDS lists (wave-synchronous staging, no barriers in
// the scan loop), then wave 0 merges the two sorted lists per row.
// d^2 via V_WMMA_F32_16X16X4_F32 with A=(gx,gy,|g|^2,1), B=(-2px,-2py,1,|p|^2)
// ---------------------------------------------------------------------------
__global__ __launch_bounds__(64)
void pn_knn_kernel(const float* __restrict__ grid,
                   const float* __restrict__ points,
                   int G, int N,
                   float* __restrict__ ws_d2,
                   int*   __restrict__ ws_idx,
                   int*   __restrict__ ws_cnt)
{
    __shared__ float stage[2][32 * 16];   // per-wave 16x16 d^2 tile pair
    __shared__ float d2l  [2][32 * 64];   // per-wave per-row sorted distances
    __shared__ int   idxl [2][32 * 64];   // per-wave per-row sorted indices
    __shared__ int   wcnt [2][32];

    const int  tid  = threadIdx.x;
    const int  wid  = tid >> 5;           // wave 0 / 1
    const int  lane = tid & 31;
    const int  base = blockIdx.x * 32;    // first grid row of this block
    const bool hi   = lane >= 16;
    const int  l16  = lane & 15;

    // A operands: tile0 = rows base..base+15, tile1 = rows base+16..base+31.
    // lo lanes carry K0,K1 = (gx,gy); hi lanes carry K2,K3 = (|g|^2, 1).
    const int gi0 = min(base + l16,      G - 1);
    const int gi1 = min(base + 16 + l16, G - 1);
    const float2 g0 = ((const float2*)grid)[gi0];
    const float2 g1 = ((const float2*)grid)[gi1];
    v2f A0, A1;
    A0[0] = hi ? (g0.x * g0.x + g0.y * g0.y) : g0.x;
    A0[1] = hi ? 1.0f : g0.y;
    A1[0] = hi ? (g1.x * g1.x + g1.y * g1.y) : g1.x;
    A1[1] = hi ? 1.0f : g1.y;

    int   cnt   = 0;
    float worst = BIGF;

    const int ntiles = (N + 15) / 16;
    const float2* pv = (const float2*)points;
    const int off = lane * 64;
    float* const stg  = &stage[wid][0];
    float* const dl   = &d2l[wid][0];
    int*   const il   = &idxl[wid][0];

    for (int pt = wid; pt < ntiles; pt += 2) {
        if (pt + 8 < ntiles)
            __builtin_prefetch(&pv[(pt + 8) * 16 + l16], 0, 3);

        // B operand: lo lanes K0,K1 = (-2px,-2py); hi lanes K2,K3 = (1,|p|^2)
        const int pid = min(pt * 16 + l16, N - 1);
        const float2 p = pv[pid];
        v2f B;
        B[0] = hi ? 1.0f : (-2.0f * p.x);
        B[1] = hi ? (p.x * p.x + p.y * p.y) : (-2.0f * p.y);

        v8f cz = {};
        v8f d0 = __builtin_amdgcn_wmma_f32_16x16x4_f32(false, A0, false, B,
                                                       (short)0, cz, false, false);
        v8f d1 = __builtin_amdgcn_wmma_f32_16x16x4_f32(false, A1, false, B,
                                                       (short)0, cz, false, false);
        // D layout: VGPR j -> lanes 0-15 hold M=j, lanes 16-31 hold M=j+8, N=l16
        #pragma unroll
        for (int jv = 0; jv < 8; ++jv) {
            const int m = jv + (hi ? 8 : 0);
            stg[m * 16 + l16]        = d0[jv];
            stg[(16 + m) * 16 + l16] = d1[jv];
        }

        // lane r owns grid row (base + r): vector load row, min-tree reject
        const float4 q0 = ((const float4*)(stg + lane * 16))[0];
        const float4 q1 = ((const float4*)(stg + lane * 16))[1];
        const float4 q2 = ((const float4*)(stg + lane * 16))[2];
        const float4 q3 = ((const float4*)(stg + lane * 16))[3];
        const float thr = fminf(R2MAX, worst);
        const float mn =
            fminf(fminf(fminf(fminf(q0.x, q0.y), fminf(q0.z, q0.w)),
                        fminf(fminf(q1.x, q1.y), fminf(q1.z, q1.w))),
                  fminf(fminf(fminf(q2.x, q2.y), fminf(q2.z, q2.w)),
                        fminf(fminf(q3.x, q3.y), fminf(q3.z, q3.w))));
        if (mn < thr) {
            const float vals[16] = { q0.x, q0.y, q0.z, q0.w,
                                     q1.x, q1.y, q1.z, q1.w,
                                     q2.x, q2.y, q2.z, q2.w,
                                     q3.x, q3.y, q3.z, q3.w };
            #pragma unroll
            for (int n = 0; n < 16; ++n) {
                const float dv = vals[n];
                const int   id = pt * 16 + n;
                if (dv < R2MAX && dv < worst && id < N) {
                    int i = (cnt < 64) ? cnt : 63;
                    while (i > 0 && dl[off + i - 1] > dv) {
                        dl[off + i] = dl[off + i - 1];
                        il[off + i] = il[off + i - 1];
                        --i;
                    }
                    dl[off + i] = dv;
                    il[off + i] = id;
                    if (cnt < 64) ++cnt;
                    if (cnt == 64) worst = dl[off + 63];
                }
            }
        }
    }

    // pad own list + publish count
    for (int i = cnt; i < 64; ++i) {
        dl[off + i] = BIGF;
        il[off + i] = 0;
    }
    wcnt[wid][lane] = cnt;
    __syncthreads();

    // wave 0: two-pointer merge of the two sorted lists per row
    if (tid < 32) {
        const int row = tid;
        if (base + row < G) {
            const int cA = wcnt[0][row];
            const int cB = wcnt[1][row];
            const int oA = row * 64;
            int ia = 0, ib = 0;
            const size_t gofs = (size_t)(base + row) * 64;
            for (int o = 0; o < 64; ++o) {         // ia+ib == o <= 63, in-range
                const float va = d2l[0][oA + ia];
                const float vb = d2l[1][oA + ib];
                const bool takeA = va <= vb;
                const float v  = takeA ? va : vb;
                const int   id = takeA ? idxl[0][oA + ia] : idxl[1][oA + ib];
                if (takeA) ++ia; else ++ib;
                ws_d2 [gofs + o] = v;
                ws_idx[gofs + o] = id;
            }
            ws_cnt[base + row] = min(64, cA + cB);
        }
    }
}

// ---------------------------------------------------------------------------
// Kernel 2: per-grid-point MLP + masked max-pool. One wave per grid point.
// Layer 1 (2->32, bias folded):  V_WMMA_F32_16X16X4_F32
// Layer 2 (32->64, bias in C):   V_WMMA_F32_16X16X32_F16 (K=32 exact fit)
// ---------------------------------------------------------------------------
__global__ __launch_bounds__(32)
void pn_mlp_kernel(const float* __restrict__ grid,
                   const float* __restrict__ points,
                   const float* __restrict__ W1,   // [3][2][32]
                   const float* __restrict__ b1,   // [3][32]
                   const float* __restrict__ W2,   // [3][32][64]
                   const float* __restrict__ b2,   // [3][64]
                   const float* __restrict__ ws_d2,
                   const int*   __restrict__ ws_idx,
                   const int*   __restrict__ ws_cnt,
                   float* __restrict__ out, int G)
{
    __shared__ _Float16 hstage[16 * 32];   // h tile: 16 neighbors x 32 features

    const int  g    = blockIdx.x;
    const int  lane = threadIdx.x;
    const bool hi   = lane >= 16;
    const int  l16  = lane & 15;

    const float2 gp  = ((const float2*)grid)[g];
    const int    cnt = ws_cnt[g];

    // sorted-prefix valid limits per scale (prefix property: list is sorted)
    const float d2a = ws_d2[(size_t)g * 64 + lane];   // padded entries = BIGF
    const int lim0 = (int)__popcll(__ballot((lane < 16) && (d2a < R2MIN)));
    const int lim1 = (int)__popcll(__ballot(d2a < R2MID));
    const int lim2 = cnt;

    for (int s = 0; s < 3; ++s) {
        // ---- preload layer-2 B tiles (W2 as f16, WMMA B layout) ----
        // lane n<16: col n, K in {0..7,16..23}; lane n>=16: col n-16, K in {8..15,24..31}
        v16h Bs[4];
        #pragma unroll
        for (int nt = 0; nt < 4; ++nt) {
            const int cn = nt * 16 + l16;
            const int kb = hi ? 8 : 0;
            #pragma unroll
            for (int kk = 0; kk < 8; ++kk) {
                Bs[nt][kk]     = (_Float16)W2[((s * 32) + kb + kk) * 64 + cn];
                Bs[nt][kk + 8] = (_Float16)W2[((s * 32) + kb + 16 + kk) * 64 + cn];
            }
        }

        // ---- layer-1 B operands (per-scale constants, hoisted) ----
        v2f Bw0, Bw1;
        {
            const int c0 = l16, c1 = 16 + l16;
            Bw0[0] = hi ? b1[s * 32 + c0] : W1[(s * 2 + 0) * 32 + c0];
            Bw0[1] = hi ? 0.0f            : W1[(s * 2 + 1) * 32 + c0];
            Bw1[0] = hi ? b1[s * 32 + c1] : W1[(s * 2 + 0) * 32 + c1];
            Bw1[1] = hi ? 0.0f            : W1[(s * 2 + 1) * 32 + c1];
        }

        const int lim = (s == 0) ? lim0 : (s == 1) ? lim1 : lim2;
        const int mt  = (s == 0) ? 1 : (s == 1) ? 2 : 4;   // M-tiles (k_s/16)

        float acc0 = -BIGF, acc1 = -BIGF, acc2 = -BIGF, acc3 = -BIGF;

        for (int t = 0; t < mt; ++t) {
            // ---- layer 1: A = rel_pos rows (lo lanes), bias constants (hi) ----
            float rx = 0.0f, ry = 0.0f;
            if (!hi) {
                const int j   = t * 16 + l16;
                const int pid = ws_idx[(size_t)g * 64 + j];
                const float2 pp = ((const float2*)points)[pid];
                rx = pp.x - gp.x;
                ry = pp.y - gp.y;
            }
            v2f A1v;
            A1v[0] = hi ? 1.0f : rx;   // K2 = 1 (bias row)
            A1v[1] = hi ? 0.0f : ry;   // K3 = 0

            __syncthreads();
            #pragma unroll
            for (int nh = 0; nh < 2; ++nh) {
                const int col = nh * 16 + l16;
                const v2f Bw = (nh == 0) ? Bw0 : Bw1;
                v8f cz = {};
                v8f h = __builtin_amdgcn_wmma_f32_16x16x4_f32(false, A1v, false, Bw,
                                                              (short)0, cz, false, false);
                #pragma unroll
                for (int jv = 0; jv < 8; ++jv) {
                    const int m = jv + (hi ? 8 : 0);
                    float hv = h[jv];
                    hv = hv > 0.0f ? hv : 0.0f;            // relu
                    hstage[m * 32 + col] = (_Float16)hv;
                }
            }
            __syncthreads();

            // ---- layer 2 A (f16, WMMA A layout) from hstage ----
            v16h Af = {};
            {
                const int row = l16;
                const int kb  = hi ? 8 : 0;
                #pragma unroll
                for (int kk = 0; kk < 8; ++kk) {
                    Af[kk]     = hstage[row * 32 + kb + kk];
                    Af[kk + 8] = hstage[row * 32 + kb + 16 + kk];
                }
            }

            #pragma unroll
            for (int nt = 0; nt < 4; ++nt) {
                const float bias = b2[s * 64 + nt * 16 + l16];
                v8f c = { bias, bias, bias, bias, bias, bias, bias, bias };
                v8f D = __builtin_amdgcn_wmma_f32_16x16x32_f16(false, Af, false, Bs[nt],
                                                               (short)0, c, false, false);
                float amax = -BIGF;
                #pragma unroll
                for (int jv = 0; jv < 8; ++jv) {
                    const int j = t * 16 + jv + (hi ? 8 : 0);
                    const float v = (j < lim) ? D[jv] : -BIGF;
                    amax = fmaxf(amax, v);
                }
                if (nt == 0) acc0 = fmaxf(acc0, amax);
                else if (nt == 1) acc1 = fmaxf(acc1, amax);
                else if (nt == 2) acc2 = fmaxf(acc2, amax);
                else acc3 = fmaxf(acc3, amax);
            }
        }

        // combine row-halves (lanes L and L+16 hold same output column) + write
        float accs[4] = { acc0, acc1, acc2, acc3 };
        #pragma unroll
        for (int nt = 0; nt < 4; ++nt) {
            float v = accs[nt];
            v = fmaxf(v, __shfl_xor(v, 16, 32));
            if (lim == 0) v = 0.0f;
            if (!hi)
                out[(size_t)g * 192 + s * 64 + nt * 16 + l16] = v;
        }
    }
}

// ---------------------------------------------------------------------------
extern "C" void kernel_launch(void* const* d_in, const int* in_sizes, int n_in,
                              void* d_out, int out_size, void* d_ws, size_t ws_size,
                              hipStream_t stream)
{
    const float* grid   = (const float*)d_in[0];
    const float* points = (const float*)d_in[1];
    const float* W1     = (const float*)d_in[2];
    const float* b1     = (const float*)d_in[3];
    const float* W2     = (const float*)d_in[4];
    const float* b2     = (const float*)d_in[5];

    const int G = in_sizes[0] / 2;
    const int N = in_sizes[1] / 2;

    float* ws_d2  = (float*)d_ws;
    int*   ws_idx = (int*)((char*)d_ws + (size_t)G * 64 * sizeof(float));
    int*   ws_cnt = (int*)((char*)d_ws + (size_t)G * 64 * (sizeof(float) + sizeof(int)));

    const int nblk1 = (G + 31) / 32;
    pn_knn_kernel<<<nblk1, 64, 0, stream>>>(grid, points, G, N, ws_d2, ws_idx, ws_cnt);
    pn_mlp_kernel<<<G, 32, 0, stream>>>(grid, points, W1, b1, W2, b2,
                                        ws_d2, ws_idx, ws_cnt, (float*)d_out, G);
}
